// SelfAttention_39874476376764
// MI455X (gfx1250) — matrix operand order repaired and checked by reading, not compiled, not run on hardware
//
#include <hip/hip_runtime.h>
#include <stdint.h>

// ---------------------------------------------------------------------------
// Flash-attention forward (causal + key padding), bf16 WMMA, f32 accumulate.
// B=2, S=2048, H=16, D=64.  qkv layout: [B, S, 3, H, D] f32.  out: [B,S,H,D] f32.
//  - 4 wave32 waves / block, each owns a 16-row query tile (64 rows / block)
//  - 64-key tiles staged in double-buffered LDS (K pre-scaled, V transposed)
//  - softmax denominator computed by the matrix pipe via an all-ones B tile
//  - exp in base-2 (log2e folded into K scale and mask constants)
//  - causal cndmasks only on the diagonal tile (wave-uniform branch)
// ---------------------------------------------------------------------------

typedef __attribute__((ext_vector_type(16))) __bf16 v16bf;
typedef __attribute__((ext_vector_type(8)))  __bf16 v8bf;
typedef __attribute__((ext_vector_type(4)))  __bf16 v4bf;
typedef __attribute__((ext_vector_type(8)))  float  v8f;

#define B_  2
#define S_  2048
#define H_  16
#define D_  64
#define LOG2E 1.44269504088896340736f
#define NEG2  (-10000.0f * LOG2E)      // reference mask value, base-2 domain

// Load 16 consecutive bf16 from LDS as one A/B fragment (2 x ds_load_b128).
static __device__ __forceinline__ v16bf lds_load16(const __bf16* p) {
  v8bf lo = *(const v8bf*)p;
  v8bf hi = *(const v8bf*)(p + 8);
  v16bf r;
#pragma unroll
  for (int i = 0; i < 8; ++i) { r[i] = lo[i]; r[8 + i] = hi[i]; }
  return r;
}

__global__ __launch_bounds__(128)
void fa_fwd_cdna5(const float* __restrict__ qkv,
                  const uint8_t* __restrict__ kpm,
                  float* __restrict__ out) {
  const int qc   = blockIdx.x;      // 64 query rows per block
  const int h    = blockIdx.y;
  const int b    = blockIdx.z;
  const int tid  = threadIdx.x;
  const int wave = tid >> 5;        // 4 waves / block (wave32)
  const int lane = tid & 31;
  const int g    = lane >> 4;       // lane half (WMMA k/m group)
  const int ln   = lane & 15;

  const int HD = H_ * D_;           // 1024
  const int RS = 3 * HD;            // 3072 floats between consecutive s
  const float scale2 = 0.125f * LOG2E;   // 1/sqrt(64) * log2(e), folded into K

  // double-buffered 64-key K/V tiles + per-wave P scratch: 40 KB total
  __shared__ __attribute__((aligned(32))) __bf16 Ks [2][64][64];   // [key][d]
  __shared__ __attribute__((aligned(32))) __bf16 VsT[2][64][64];   // [d][key]
  __shared__ __attribute__((aligned(32))) __bf16 Pb [4][16][64];   // per-wave P

  const size_t bh = (size_t)b * S_ * RS + (size_t)h * D_;
  const float* qp = qkv + bh;            // Q row t: + t*RS
  const float* kp = qkv + bh + HD;       // K
  const float* vp = qkv + bh + 2 * HD;   // V
  const uint8_t* kpmb = kpm + (size_t)b * S_;

  const int qbase = qc * 64 + wave * 16;
  const int qlast = qbase + 15;

  // ---- Q A-fragments (16x32 bf16; element e: d = g*16 + e) ----
  const float* qrow = qp + (size_t)(qbase + ln) * RS;
  v16bf a0, a1;
#pragma unroll
  for (int hh = 0; hh < 2; ++hh) {
#pragma unroll
    for (int j = 0; j < 8; ++j) {
      const int d0 = g * 16 + hh * 8 + j;
      a0[hh * 8 + j] = (__bf16)qrow[d0];
      a1[hh * 8 + j] = (__bf16)qrow[32 + d0];
    }
  }

  // all-ones B fragment: P x ones accumulates the softmax denominator
  v16bf ones;
#pragma unroll
  for (int i = 0; i < 16; ++i) ones[i] = (__bf16)1.0f;

  // ---- online-softmax state (C-layout: slot r <-> row m = r + 8*g) ----
  float mrow[8];
  v8f acc[4];                        // out tile 16x64 = 4 x (16x16 f32)
  v8f lfrag;                         // row-sum accumulator (all columns equal)
  {
    v8f z = {};
#pragma unroll
    for (int f = 0; f < 4; ++f) acc[f] = z;
    lfrag = z;
#pragma unroll
    for (int r = 0; r < 8; ++r) mrow[r] = -INFINITY;
  }

  const int ktmax = qc + 1;          // 64-key tiles; last touches row qc*64+63
  for (int kt = 0; kt < ktmax; ++kt) {
    const int kbase = kt * 64;
    const int buf   = kt & 1;

    // ---- cooperative staging: K (scaled) row-major, V transposed ----
#pragma unroll
    for (int i = 0; i < 8; ++i) {
      const int c   = tid + i * 128;     // 1024 float4 chunks
      const int row = c >> 4;            // key 0..63
      const int col = (c & 15) * 4;      // d
      const float4 kf = *(const float4*)(kp + (size_t)(kbase + row) * RS + col);
      v4bf kb;
      kb[0] = (__bf16)(kf.x * scale2); kb[1] = (__bf16)(kf.y * scale2);
      kb[2] = (__bf16)(kf.z * scale2); kb[3] = (__bf16)(kf.w * scale2);
      *(v4bf*)&Ks[buf][row][col] = kb;
      const float4 vf = *(const float4*)(vp + (size_t)(kbase + row) * RS + col);
      VsT[buf][col + 0][row] = (__bf16)vf.x; VsT[buf][col + 1][row] = (__bf16)vf.y;
      VsT[buf][col + 2][row] = (__bf16)vf.z; VsT[buf][col + 3][row] = (__bf16)vf.w;
    }
    if (kt + 1 < ktmax) {              // speculative prefetch of next tile
      const int pr = tid >> 1, pc = (tid & 1) * 32;   // 64 rows x 2 half-rows
      __builtin_prefetch(kp + (size_t)(kbase + 64 + pr) * RS + pc, 0, 1);
      __builtin_prefetch(vp + (size_t)(kbase + 64 + pr) * RS + pc, 0, 1);
    }
    __syncthreads();                   // single barrier/iter (double-buffered)

    if (kbase <= qlast) {              // wave-uniform: EXEC all-ones inside
      // ---- scores: four 16x16 tiles, D=64 contracted in two K=32 WMMAs ----
      v8f c4[4];
#pragma unroll
      for (int nt = 0; nt < 4; ++nt) {
        v16bf kb0 = lds_load16(&Ks[buf][nt * 16 + ln][g * 16]);
        v16bf kb1 = lds_load16(&Ks[buf][nt * 16 + ln][32 + g * 16]);
        v8f z = {};
        v8f cc = __builtin_amdgcn_wmma_f32_16x16x32_bf16(false, a0, false, kb0, (short)0, z,  false, false);
        c4[nt]  = __builtin_amdgcn_wmma_f32_16x16x32_bf16(false, a1, false, kb1, (short)0, cc, false, false);
      }

      float pad[4];
#pragma unroll
      for (int nt = 0; nt < 4; ++nt)
        pad[nt] = kpmb[kbase + nt * 16 + ln] ? 0.0f : NEG2;
      const bool diag = (kbase + 63) > qbase;   // only diagonal tile masks

      float alpha[8];
#pragma unroll
      for (int r = 0; r < 8; ++r) {
        const int t = qbase + r + 8 * g;
#pragma unroll
        for (int nt = 0; nt < 4; ++nt) {
          float x = c4[nt][r] + pad[nt];
          if (diag)                                // wave-uniform branch
            x += ((kbase + nt * 16 + ln) > t) ? NEG2 : 0.0f;
          c4[nt][r] = x;
        }
        // row-max over 64 keys: 3 local fmax + one 16-lane reduction
        float mx = fmaxf(fmaxf(c4[0][r], c4[1][r]), fmaxf(c4[2][r], c4[3][r]));
        mx = fmaxf(mx, __shfl_xor(mx, 1, 32));
        mx = fmaxf(mx, __shfl_xor(mx, 2, 32));
        mx = fmaxf(mx, __shfl_xor(mx, 4, 32));
        mx = fmaxf(mx, __shfl_xor(mx, 8, 32));
        const float mnew = fmaxf(mrow[r], mx);
        alpha[r] = __builtin_amdgcn_exp2f(mrow[r] - mnew);  // exp2(-inf)=0 first
        mrow[r]  = mnew;
        // exp + C-layout -> row-major bf16 P in per-wave LDS
        const int m = r + 8 * g;
#pragma unroll
        for (int nt = 0; nt < 4; ++nt)
          Pb[wave][m][nt * 16 + ln] =
              (__bf16)__builtin_amdgcn_exp2f(c4[nt][r] - mnew);
      }
#pragma unroll
      for (int f = 0; f < 4; ++f)
#pragma unroll
        for (int r = 0; r < 8; ++r) acc[f][r] *= alpha[r];
#pragma unroll
      for (int r = 0; r < 8; ++r) lfrag[r] *= alpha[r];

      asm volatile("s_wait_dscnt 0x0" ::: "memory");   // order store->gather in-wave

      // ---- gather P as two A-fragments; PV + denominator WMMAs ----
#pragma unroll
      for (int half = 0; half < 2; ++half) {
        v16bf pa;
        {
          v8bf lo = *(const v8bf*)&Pb[wave][ln][half * 32 + g * 16];
          v8bf hi = *(const v8bf*)&Pb[wave][ln][half * 32 + g * 16 + 8];
#pragma unroll
          for (int i = 0; i < 8; ++i) { pa[i] = lo[i]; pa[8 + i] = hi[i]; }
        }
#pragma unroll
        for (int f = 0; f < 4; ++f) {
          v16bf vb = lds_load16(&VsT[buf][f * 16 + ln][half * 32 + g * 16]);
          acc[f] = __builtin_amdgcn_wmma_f32_16x16x32_bf16(false, pa, false, vb, (short)0, acc[f], false, false);
        }
        lfrag = __builtin_amdgcn_wmma_f32_16x16x32_bf16(false, pa, false, ones, (short)0, lfrag, false, false);
      }
    }
  }

  // ---- normalize and write out[b][t][h][d] ----
  float inv[8];
#pragma unroll
  for (int r = 0; r < 8; ++r) inv[r] = (lfrag[r] > 0.0f) ? (1.0f / lfrag[r]) : 0.0f;
#pragma unroll
  for (int f = 0; f < 4; ++f) {
#pragma unroll
    for (int r = 0; r < 8; ++r) {
      const int t = qbase + r + 8 * g;
      out[(((size_t)b * S_ + t) * H_ + h) * D_ + f * 16 + ln] = acc[f][r] * inv[r];
    }
  }
}

extern "C" void kernel_launch(void* const* d_in, const int* in_sizes, int n_in,
                              void* d_out, int out_size, void* d_ws, size_t ws_size,
                              hipStream_t stream) {
  (void)in_sizes; (void)n_in; (void)out_size; (void)d_ws; (void)ws_size;
  const float*   qkv = (const float*)d_in[0];
  const uint8_t* kpm = (const uint8_t*)d_in[1];   // jnp bool -> 1 byte/elem
  float*         out = (float*)d_out;
  dim3 grid(S_ / 64, H_, B_);   // (query chunks, heads, batch) = (32,16,2)
  dim3 block(128);              // 4 wave32 waves
  fa_fwd_cdna5<<<grid, block, 0, stream>>>(qkv, kpm, out);
}
